// Qwen3VLMoeTextExperts_37812892074071
// MI455X (gfx1250) — matrix operand order repaired and checked by reading, not compiled, not run on hardware
//
#include <hip/hip_runtime.h>
#include <math.h>

// Problem constants (match reference)
#define NE   32      // experts
#define ND   2048    // hidden size D
#define NF   768     // intermediate F
#define TWOF 1536    // 2F
#define NT   2048    // tokens T = B*S

#define BM   64      // tokens per block
// gated LDS row stride in bf16 elements (768 + 8 pad; 776*2=1552 bytes, 16B aligned)
#define GS   776

typedef __attribute__((ext_vector_type(16))) __bf16 v16bf;
typedef __attribute__((ext_vector_type(8)))  __bf16 v8bf;
typedef __attribute__((ext_vector_type(8)))  float  v8f;

static __device__ __forceinline__ v16bf ld16(const __bf16* p0, const __bf16* p1) {
    v8bf lo = *(const v8bf*)p0;
    v8bf hi = *(const v8bf*)p1;
    v16bf r;
#pragma unroll
    for (int i = 0; i < 8; ++i) { r[i] = lo[i]; r[i + 8] = hi[i]; }
    return r;
}

static __device__ __forceinline__ v8f wmma_bf16(v16bf a, v16bf b, v8f c) {
    // D = A(16x32 bf16) * B(32x16 bf16) + C(16x16 f32)
    return __builtin_amdgcn_wmma_f32_16x16x32_bf16(false, a, false, b, (short)0, c,
                                                   false, false);
}

// Load 8 K-consecutive f32 (srcStride elements between K rows), convert to bf16,
// store as ONE 16-byte LDS write into the transposed [n][k] slab. The 8 global
// dword loads are fully coalesced across lanes (n is the lane-varying index).
static __device__ __forceinline__ void stageK8(__bf16* dst, const float* src, int srcStride) {
    float f0 = src[0 * (size_t)srcStride];
    float f1 = src[1 * (size_t)srcStride];
    float f2 = src[2 * (size_t)srcStride];
    float f3 = src[3 * (size_t)srcStride];
    float f4 = src[4 * (size_t)srcStride];
    float f5 = src[5 * (size_t)srcStride];
    float f6 = src[6 * (size_t)srcStride];
    float f7 = src[7 * (size_t)srcStride];
    v8bf pk;
    pk[0] = (__bf16)f0; pk[1] = (__bf16)f1; pk[2] = (__bf16)f2; pk[3] = (__bf16)f3;
    pk[4] = (__bf16)f4; pk[5] = (__bf16)f5; pk[6] = (__bf16)f6; pk[7] = (__bf16)f7;
    *(v8bf*)dst = pk;
}

__global__ void zero_out_kernel(float4* __restrict__ out, int n4) {
    int i = blockIdx.x * blockDim.x + threadIdx.x;
    if (i < n4) out[i] = make_float4(0.f, 0.f, 0.f, 0.f);
}

__global__ __launch_bounds__(256)
void moe_experts_kernel(const float* __restrict__ x,     // (T, D)
                        const float* __restrict__ rwt,   // (T, E)
                        const float* __restrict__ gup,   // (E, D, 2F)
                        const float* __restrict__ dwn,   // (E, F, D)
                        float* __restrict__ out)         // (T, D), pre-zeroed
{
    // 123.9 KB static LDS (gfx1250 WGP has 320 KB)
    __shared__ __align__(16) __bf16 gatedLds[BM * GS];      // 99,328 B : gated 64x768 bf16
    __shared__ __align__(16) __bf16 Xs[2 * BM * 32];        //  8 KB    : double-buffered x slab [row][k]
    __shared__ __align__(16) __bf16 Wslab[2 * 128 * 32];    // 16 KB    : double-buffered weight slab [n][k]

    const int e  = blockIdx.y;
    const int t0 = blockIdx.x * BM;     // 64-token tile

    const int tid  = threadIdx.x;
    const int lane = tid & 31;
    const int wv   = tid >> 5;          // wave 0..7
    const int mt   = wv >> 1;           // 16-row M tile (0..3)
    const int nq   = wv & 1;            // N slot (0..1)
    const int ln   = lane & 15;
    const int lg   = lane >> 4;

    const float* gupE = gup + (size_t)e * ND * TWOF;
    const float* dwnE = dwn + (size_t)e * NF * ND;

    // routing weights for the 8 token rows this lane owns in any C tile
    // C layout: N = lane&15, M = v + 8*(lane>>4)
    float rwl[8];
#pragma unroll
    for (int v = 0; v < 8; ++v) {
        int tok = t0 + mt * 16 + lg * 8 + v;
        rwl[v] = rwt[(size_t)tok * NE + e];
    }

    // ---- staging helpers (each a full-block cooperative copy, no barrier inside) ----
    // X slab: BM x 32 f32 -> bf16, row-major [row][k]; 8 k per thread, 1 b128 LDS store.
    auto stage_x = [&](int k0, int buf) {
        int i0 = tid * 8;                       // 2048 elems / 256 thr
        int r = i0 >> 5, k = i0 & 31;
        const float* src = x + (size_t)(t0 + r) * ND + k0 + k;
        float4 f0 = *(const float4*)src;
        float4 f1 = *(const float4*)(src + 4);
        v8bf pk;
        pk[0] = (__bf16)f0.x; pk[1] = (__bf16)f0.y; pk[2] = (__bf16)f0.z; pk[3] = (__bf16)f0.w;
        pk[4] = (__bf16)f1.x; pk[5] = (__bf16)f1.y; pk[6] = (__bf16)f1.z; pk[7] = (__bf16)f1.w;
        *(v8bf*)&Xs[buf * (BM * 32) + r * 32 + k] = pk;
    };
    // GEMM1 weight slabs: gate 32k x 64n at +0, up 32k x 64n at +2048, layout [n][k]
    auto stage_w1 = [&](int c0, int k0, int buf) {
        int n    = tid & 63;
        int half = (tid >> 6) & 1;              // 0 = gate, 1 = up
        __bf16* base = Wslab + buf * 4096 + half * 2048;
        const float* gsrc = gupE + (size_t)k0 * TWOF + half * NF + c0 + n;
#pragma unroll
        for (int it = 0; it < 2; ++it) {
            int kb = (((tid >> 7) & 1) + 2 * it) * 8;   // 0,8,16,24
            stageK8(&base[n * 32 + kb], gsrc + (size_t)kb * TWOF, TWOF);
        }
        if (k0 + 64 < ND)   // L2 prefetch two K-slabs ahead
            __builtin_prefetch(gupE + (size_t)(k0 + 64) * TWOF + c0 + n, 0, 1);
    };
    // GEMM2 weight slab: 32k x 128n, layout [n][k]
    auto stage_w2 = [&](int d0, int k0, int buf) {
        int n = tid & 127;
        __bf16* base = Wslab + buf * 4096;
        const float* dsrc = dwnE + (size_t)k0 * ND + d0 + n;
#pragma unroll
        for (int it = 0; it < 2; ++it) {
            int kb = (((tid >> 7) & 1) + 2 * it) * 8;   // 0,8,16,24
            stageK8(&base[n * 32 + kb], dsrc + (size_t)kb * ND, ND);
        }
    };

    // ---------------- GEMM1 (x @ W1) + SiLU + route-weight fold ----------------
    // Wave (mt, nq) owns rows [mt*16, +16) and cols [c0 + nq*32, +32): 2 gate + 2 up tiles.
    for (int ch = 0; ch < NF / 64; ++ch) {      // 12 chunks of 64 F-cols
        const int c0 = ch * 64;
        v8f aG0 = {}, aG1 = {}, aU0 = {}, aU1 = {};
        stage_x(0, 0);
        stage_w1(c0, 0, 0);
#pragma unroll 2
        for (int ks = 0; ks < ND / 32; ++ks) {  // 64 K-steps, double buffered, 1 barrier each
            const int cur = ks & 1;
            __syncthreads();
            if (ks + 1 < ND / 32) {             // stage next step while computing current
                stage_x((ks + 1) * 32, cur ^ 1);
                stage_w1(c0, (ks + 1) * 32, cur ^ 1);
            }
            const __bf16* Xb = Xs + cur * (BM * 32);
            const __bf16* pa = Xb + (mt * 16 + ln) * 32 + lg * 8;
            v16bf A = ld16(pa, pa + 16);
            const __bf16* Wb = Wslab + cur * 4096;
            const __bf16* g0 = Wb + (nq * 32 + ln) * 32 + lg * 8;
            const __bf16* g1 = Wb + (nq * 32 + 16 + ln) * 32 + lg * 8;
            const __bf16* u0 = Wb + 2048 + (nq * 32 + ln) * 32 + lg * 8;
            const __bf16* u1 = Wb + 2048 + (nq * 32 + 16 + ln) * 32 + lg * 8;
            aG0 = wmma_bf16(A, ld16(g0, g0 + 16), aG0);
            aG1 = wmma_bf16(A, ld16(g1, g1 + 16), aG1);
            aU0 = wmma_bf16(A, ld16(u0, u0 + 16), aU0);
            aU1 = wmma_bf16(A, ld16(u1, u1 + 16), aU1);
        }
        // gated = up * silu(gate) * routing_weight  -> bf16 LDS
#pragma unroll
        for (int v = 0; v < 8; ++v) {
            int r = mt * 16 + lg * 8 + v;
            float s0 = aG0[v] / (1.0f + __expf(-aG0[v]));
            float s1 = aG1[v] / (1.0f + __expf(-aG1[v]));
            gatedLds[r * GS + c0 + nq * 32 + ln]      = (__bf16)(aU0[v] * s0 * rwl[v]);
            gatedLds[r * GS + c0 + nq * 32 + 16 + ln] = (__bf16)(aU1[v] * s1 * rwl[v]);
        }
    }
    __syncthreads();   // gated complete; weight slab free for GEMM2

    // ------------- GEMM2 (gated @ W2), 128-col chunks, atomic accumulate -------------
    // Wave (mt, nq) owns rows [mt*16,+16) and cols [d0 + nq*64, +64): 4 tiles.
    for (int ch = 0; ch < ND / 128; ++ch) {     // 16 chunks
        const int d0 = ch * 128;
        v8f a0 = {}, a1 = {}, a2 = {}, a3 = {};
        stage_w2(d0, 0, 0);
#pragma unroll 2
        for (int ks = 0; ks < NF / 32; ++ks) {  // 24 K-steps, double buffered
            const int cur = ks & 1;
            __syncthreads();
            if (ks + 1 < NF / 32)
                stage_w2(d0, (ks + 1) * 32, cur ^ 1);
            const __bf16* pa = gatedLds + (mt * 16 + ln) * GS + ks * 32 + lg * 8;
            v16bf A = ld16(pa, pa + 16);
            const __bf16* Wb = Wslab + cur * 4096;
            const __bf16* b0 = Wb + (nq * 64 + ln) * 32 + lg * 8;
            const __bf16* b1 = Wb + (nq * 64 + 16 + ln) * 32 + lg * 8;
            const __bf16* b2 = Wb + (nq * 64 + 32 + ln) * 32 + lg * 8;
            const __bf16* b3 = Wb + (nq * 64 + 48 + ln) * 32 + lg * 8;
            a0 = wmma_bf16(A, ld16(b0, b0 + 16), a0);
            a1 = wmma_bf16(A, ld16(b1, b1 + 16), a1);
            a2 = wmma_bf16(A, ld16(b2, b2 + 16), a2);
            a3 = wmma_bf16(A, ld16(b3, b3 + 16), a3);
        }
#pragma unroll
        for (int v = 0; v < 8; ++v) {
            int tok = t0 + mt * 16 + lg * 8 + v;
            float* po = &out[(size_t)tok * ND + d0 + nq * 64 + ln];
            atomicAdd(po,      a0[v]);
            atomicAdd(po + 16, a1[v]);
            atomicAdd(po + 32, a2[v]);
            atomicAdd(po + 48, a3[v]);
        }
    }
}

extern "C" void kernel_launch(void* const* d_in, const int* in_sizes, int n_in,
                              void* d_out, int out_size, void* d_ws, size_t ws_size,
                              hipStream_t stream) {
    const float* x    = (const float*)d_in[0];   // hidden_states (B,S,D) -> (T,D)
    const float* rwt  = (const float*)d_in[1];   // routing_weights (T,E)
    // d_in[2] = router_indices: unused by the reference
    const float* gup  = (const float*)d_in[3];   // gate_up_proj (E,D,2F)
    const float* dwn  = (const float*)d_in[4];   // down_proj    (E,F,D)
    float* out        = (float*)d_out;           // (T,D)

    // zero output (harness poisons d_out; we accumulate with atomics)
    int n4 = (NT * ND) / 4;
    zero_out_kernel<<<(n4 + 255) / 256, 256, 0, stream>>>((float4*)out, n4);

    dim3 grid(NT / BM, NE);   // 32 token tiles x 32 experts
    moe_experts_kernel<<<grid, 256, 0, stream>>>(x, rwt, gup, dwn, out);
}
